// LSHScaledDotProduct_19164144075427
// MI455X (gfx1250) — compile-verified, matrix operand order checked by reading, not run on hardware
//
#include <hip/hip_runtime.h>

// ---- problem constants (match reference) ----
#define HEADS   8
#define DK      64
#define DV      64
#define DM      (HEADS * DK)          // 512
#define ROUNDS  2
#define SEQ     4096
#define BUCKET  64
#define NCHUNK  (SEQ / BUCKET)        // 64
#define NBUCKETS (SEQ / BUCKET)       // 64
#define NPROJ   (NBUCKETS / 2)        // 32
#define WIN     (2 * BUCKET)          // 128
#define BATCHN  4
#define SCALE_INV 0.125f
#define NEG_BIG  (-1.0e15f)

typedef __attribute__((ext_vector_type(16))) __bf16 v16bf;
typedef __attribute__((ext_vector_type(8)))  float  v8f;

// ------------------------------------------------------------------
// Kernel 1: LSH hashing.  One wave32 per (r,b,s); lane = projection m.
// rot[m] = sum_d q[b,s,d] * R[r,d,m]; h = argmax([rot,-rot]) + 1.
// ------------------------------------------------------------------
__global__ __launch_bounds__(128) void lsh_hash_kernel(const float* __restrict__ q,
                                                       const float* __restrict__ R,
                                                       int* __restrict__ hout) {
    int gw   = blockIdx.x * 4 + (threadIdx.x >> 5);    // 0 .. ROUNDS*BATCHN*SEQ-1
    int lane = threadIdx.x & 31;
    int s  = gw & (SEQ - 1);
    int t  = gw >> 12;            // r*BATCHN + b
    int b  = t & (BATCHN - 1);
    int r  = t >> 2;

    const float* qr = q + ((size_t)b * SEQ + s) * DM;
    const float* Rp = R + (size_t)r * DM * NPROJ + lane;

    float acc = 0.0f;
    #pragma unroll 8
    for (int d = 0; d < DM; ++d)
        acc += qr[d] * Rp[(size_t)d * NPROJ];

    // candidates: rot (idx=lane) then -rot (idx=lane+32); ties -> first (rot)
    float v  = acc;   int idx = lane;
    float nv = -acc;
    if (nv > v) { v = nv; idx = lane + 32; }

    // wave-wide argmax, stable tie-break on smaller index
    #pragma unroll
    for (int mask = 16; mask >= 1; mask >>= 1) {
        float ov = __shfl_xor(v,   mask, 32);
        int   oi = __shfl_xor(idx, mask, 32);
        if (ov > v || (ov == v && oi < idx)) { v = ov; idx = oi; }
    }
    if (lane == 0) hout[gw] = idx + 1;     // hash in 1..64
}

// ------------------------------------------------------------------
// Kernel 2: stable counting sort by hash (keys 1..64).
// One block per (r,b), 64 threads: thread t owns key t+1.
// ------------------------------------------------------------------
__global__ __launch_bounds__(64) void lsh_sort_kernel(const int* __restrict__ h,
                                                      int* __restrict__ sidx,
                                                      int* __restrict__ hsort) {
    __shared__ int hl[SEQ];
    __shared__ int counts[NBUCKETS];
    int rb = blockIdx.x;
    const int* hp = h + (size_t)rb * SEQ;
    for (int i = threadIdx.x; i < SEQ; i += blockDim.x) hl[i] = hp[i];
    __syncthreads();

    int t   = threadIdx.x;     // 0..63
    int key = t + 1;
    int c = 0;
    for (int i = 0; i < SEQ; ++i) c += (hl[i] == key);
    counts[t] = c;
    __syncthreads();
    if (t == 0) {              // tiny exclusive scan over 64 entries
        int run = 0;
        for (int k2 = 0; k2 < NBUCKETS; ++k2) { int cc = counts[k2]; counts[k2] = run; run += cc; }
    }
    __syncthreads();
    int off = counts[t];
    for (int i = 0; i < SEQ; ++i) {
        if (hl[i] == key) {
            sidx [(size_t)rb * SEQ + off] = i;
            hsort[(size_t)rb * SEQ + off] = key;
            ++off;
        }
    }
}

// ------------------------------------------------------------------
// Kernel 3: zero the output (rounds accumulate with atomics).
// ------------------------------------------------------------------
__global__ void lsh_zero_kernel(float* __restrict__ o, int n) {
    int i = blockIdx.x * blockDim.x + threadIdx.x;
    if (i < n) o[i] = 0.0f;
}

// ------------------------------------------------------------------
// WMMA fragment helper (16-bit A/B 16x32 layout, CDNA5 ISA 7.12.2):
// lane<16 holds K in {0..7, 16..23}; lane>=16 holds K in {8..15, 24..31}.
// k0 already contains the lane-half offset (0 or 8); reads are contiguous
// per 8 elements so they vectorize to ds_load_b128 pairs.
// ------------------------------------------------------------------
static __device__ inline v16bf load_row_frag(const __bf16* rowPtr, int k0) {
    v16bf f;
    #pragma unroll
    for (int j = 0; j < 8; ++j)  f[j] = rowPtr[k0 + j];
    #pragma unroll
    for (int j = 8; j < 16; ++j) f[j] = rowPtr[k0 + j + 8];
    return f;
}

// ------------------------------------------------------------------
// Kernel 4: windowed LSH attention for one (round, batch, chunk, head).
// 128 threads = 4 waves; wave w owns query columns [16w, 16w+16).
// GEMM1 computes S^T = K * Q^T so that each query's 128 scores live in
// a single lane pair (lane q: keys rh=0, lane q+16: keys rh=1) -> the
// whole softmax needs only 2 cross-lane shuffles per lane.
// GEMM2 computes O^T = V^T * P^T; the P^T B-fragment is exactly the
// register layout the probabilities are already in (no LDS round-trip).
// ------------------------------------------------------------------
__global__ __launch_bounds__(128) void lsh_attn_kernel(const float* __restrict__ q,
                                                       const float* __restrict__ k,
                                                       const float* __restrict__ v,
                                                       const int* __restrict__ sidx,
                                                       const int* __restrict__ hsort,
                                                       float* __restrict__ out) {
    int bid = blockIdx.x;
    int hh  = bid & (HEADS - 1);  bid >>= 3;
    int n   = bid & (NCHUNK - 1); bid >>= 6;
    int b   = bid & (BATCHN - 1); bid >>= 2;
    int r   = bid;

    int rb = r * BATCHN + b;
    const int* sp = sidx  + (size_t)rb * SEQ;
    const int* hp = hsort + (size_t)rb * SEQ;

    __shared__ __bf16 Qs[BUCKET * DK];          //  8 KB : Q rows (row-major)
    __shared__ __bf16 Ks[WIN * DK];             // 16 KB : K window rows (row-major)
    __shared__ __bf16 Vt[DV * WIN];             // 16 KB : V window TRANSPOSED Vt[d][w]
    __shared__ unsigned long long hw64s[WIN/8]; // window hashes packed as bytes
    __shared__ int cntk[NBUCKETS];              // histogram of window hashes

    unsigned char* hwb = (unsigned char*)hw64s;
    int tid = threadIdx.x;
    int pn  = (n + NCHUNK - 1) & (NCHUNK - 1);  // rolled previous chunk

    if (tid < NBUCKETS) cntk[tid] = 0;

    // ---- gather: K/V window rows (row = tid), hashes, Q rows (2 thr/row) ----
    {
        int c   = (tid < BUCKET) ? pn : n;
        int p   = c * BUCKET + (tid & (BUCKET - 1));
        int src = sp[p];
        hwb[tid] = (unsigned char)hp[p];
        const float* kr = k + ((size_t)b * SEQ + src) * DM + hh * DK;
        const float* vr = v + ((size_t)b * SEQ + src) * DM + hh * DV;
        #pragma unroll 8
        for (int d = 0; d < DK; ++d) Ks[tid * DK + d]  = (__bf16)kr[d];
        #pragma unroll 8
        for (int d = 0; d < DV; ++d) Vt[d * WIN + tid] = (__bf16)vr[d];

        int qrow = tid >> 1;
        int d0   = (tid & 1) * 32;
        int src2 = sp[n * BUCKET + qrow];
        const float* qr = q + ((size_t)b * SEQ + src2) * DM + hh * DK + d0;
        #pragma unroll 8
        for (int d = 0; d < 32; ++d) Qs[qrow * DK + d0 + d] = (__bf16)qr[d];
    }
    __syncthreads();

    // histogram of window hashes (keys 1..64), one LDS atomic per thread
    atomicAdd(&cntk[(int)hwb[tid] - 1], 1);

    int lane = tid & 31;
    int wv   = tid >> 5;          // query strip 0..3
    int ln15 = lane & 15;
    int rh   = lane >> 4;         // lane half: key sub-rows / K offset
    int kb   = rh * 8;
    int m    = wv * 16 + ln15;    // this lane's query (local 0..63)

    // ---- GEMM1: S^T = K (128x64) x Q^T, per-wave N = 16 queries ----
    // acc[t] element i  ->  key l = t*16 + rh*8 + i, query = m
    v8f acc[8];
    #pragma unroll
    for (int t = 0; t < 8; ++t) acc[t] = (v8f)(0.0f);

    #pragma unroll
    for (int ks = 0; ks < 2; ++ks) {
        int k0 = ks * 32 + kb;
        v16bf bfrag = load_row_frag(&Qs[m * DK], k0);      // B column = Q row m
        #pragma unroll
        for (int t = 0; t < 8; ++t) {
            v16bf afrag = load_row_frag(&Ks[(t * 16 + ln15) * DK], k0);
            acc[t] = __builtin_amdgcn_wmma_f32_16x16x32_bf16(
                false, afrag, false, bfrag, (short)0, acc[t], false, false);
        }
    }
    __syncthreads();   // cntk complete, hashes stable

    // ---- bias + softmax: each lane owns 64 of its query's 128 scores ----
    int   hq  = (int)hwb[BUCKET + m];
    int   cnt = cntk[hq - 1];
    float nlc = -__logf((float)(cnt < 1 ? 1 : cnt));
    int   dl  = m + BUCKET;       // diagonal key index for this query

    float mx = -3.0e38f;
    #pragma unroll
    for (int t = 0; t < 8; ++t) {
        unsigned long long kk = hw64s[t * 2 + rh];   // 8 key hashes at once
        #pragma unroll
        for (int i = 0; i < 8; ++i) {
            int l  = t * 16 + kb + i;
            int hk = (int)((kk >> (8 * i)) & 0xff);
            float val = acc[t][i] * SCALE_INV + ((hk == hq) ? nlc : NEG_BIG);
            if (l == dl) val -= 100000.0f;           // eye(128) * -1e5
            acc[t][i] = val;
            mx = fmaxf(mx, val);
        }
    }
    mx = fmaxf(mx, __shfl_xor(mx, 16, 32));          // join the lane pair

    float sum = 0.0f;
    #pragma unroll
    for (int t = 0; t < 8; ++t)
        #pragma unroll
        for (int i = 0; i < 8; ++i) {
            float e = __expf(acc[t][i] - mx);
            acc[t][i] = e;
            sum += e;
        }
    sum += __shfl_xor(sum, 16, 32);
    float inv = 1.0f / sum;
    #pragma unroll
    for (int t = 0; t < 8; ++t)
        #pragma unroll
        for (int i = 0; i < 8; ++i) acc[t][i] *= inv;

    // ---- GEMM2: O^T = V^T (64x128) x P^T (128 x 16 queries) ----
    // P^T B-fragment for K-step ks comes straight from acc tiles 2ks, 2ks+1.
    v8f oacc[4];
    #pragma unroll
    for (int mt = 0; mt < 4; ++mt) oacc[mt] = (v8f)(0.0f);

    #pragma unroll
    for (int ks = 0; ks < 4; ++ks) {
        v16bf bfrag;
        #pragma unroll
        for (int j = 0; j < 8; ++j) bfrag[j]     = (__bf16)acc[2 * ks][j];
        #pragma unroll
        for (int j = 0; j < 8; ++j) bfrag[j + 8] = (__bf16)acc[2 * ks + 1][j];
        int k0 = ks * 32 + kb;
        #pragma unroll
        for (int mt = 0; mt < 4; ++mt) {
            v16bf afrag = load_row_frag(&Vt[(mt * 16 + ln15) * WIN], k0);
            oacc[mt] = __builtin_amdgcn_wmma_f32_16x16x32_bf16(
                false, afrag, false, bfrag, (short)0, oacc[mt], false, false);
        }
    }

    // ---- scatter this query's 64 output dims back to original position ----
    int orig = sp[n * BUCKET + m];
    float* op = out + (((size_t)b * SEQ + orig) * HEADS + hh) * DV;
    #pragma unroll
    for (int mt = 0; mt < 4; ++mt)
        #pragma unroll
        for (int i = 0; i < 8; ++i)
            atomicAdd(&op[mt * 16 + kb + i], oacc[mt][i]);   // dv = mt*16 + rh*8 + i
}

// ------------------------------------------------------------------
extern "C" void kernel_launch(void* const* d_in, const int* in_sizes, int n_in,
                              void* d_out, int out_size, void* d_ws, size_t ws_size,
                              hipStream_t stream) {
    const float* q = (const float*)d_in[0];
    const float* k = (const float*)d_in[1];
    const float* v = (const float*)d_in[2];
    const float* R = (const float*)d_in[3];
    float* out = (float*)d_out;

    const size_t NITEMS = (size_t)ROUNDS * BATCHN * SEQ;   // 32768
    int* h_arr = (int*)d_ws;
    int* sidx  = h_arr + NITEMS;
    int* hsort = sidx  + NITEMS;

    // 1) hashes: one wave per item, 4 waves/block
    lsh_hash_kernel<<<(int)(NITEMS / 4), 128, 0, stream>>>(q, R, h_arr);

    // 2) stable counting sort per (round, batch)
    lsh_sort_kernel<<<ROUNDS * BATCHN, 64, 0, stream>>>(h_arr, sidx, hsort);

    // 3) zero the accumulator output
    lsh_zero_kernel<<<(out_size + 255) / 256, 256, 0, stream>>>(out, out_size);

    // 4) windowed attention: one block per (r, b, chunk, head)
    lsh_attn_kernel<<<ROUNDS * BATCHN * NCHUNK * HEADS, 128, 0, stream>>>(
        q, k, v, sidx, hsort, out);
}
